// CrossSymmetricModal_40742059770051
// MI455X (gfx1250) — matrix operand order repaired and checked by reading, not compiled, not run on hardware
//
#include <hip/hip_runtime.h>
#include <hip/hip_bf16.h>

// ---------------------------------------------------------------- constants
#define Bn   16
#define Cc   256
#define Ll   1024
#define LP   1026            // L + 2 pad rows for K=3 conv
#define NROW 16384           // Bn * Ll
#define EPSV 1e-5f

#define BSTR 272             // LDS row stride (elements) for 256-wide tiles (+32B TDM pad)
#define CTX_CH 128           // ctx K-chunk (elements)
#define CTX_STR 144          // LDS row stride for 128-wide tiles (+32B TDM pad)

typedef __bf16 bf16;
typedef __attribute__((ext_vector_type(16))) __bf16 v16bf;
typedef __attribute__((ext_vector_type(8)))  __bf16 v8bf;
typedef __attribute__((ext_vector_type(8)))  float  v8f;
typedef __attribute__((ext_vector_type(4)))  unsigned int u32x4;
typedef __attribute__((ext_vector_type(8)))  int i32x8;
typedef __attribute__((ext_vector_type(4)))  int i32x4;

// ---------------------------------------------------------------- TDM
// Build D# per CDNA5 ISA ch.8 and issue TENSOR_LOAD_TO_LDS (2D tile, bf16).
// pad_iv: pad_interval code (6 => 128 DWORDs = 512B rows, 5 => 64 DWORDs = 256B rows)
// pad amount fixed at code 7 => 8 DWORDs (32B) per row -> LDS row skew.
__device__ __forceinline__ void tdm_load_2d(unsigned lds_off, const void* gptr,
                                            unsigned tile_d0, unsigned tile_d1,
                                            unsigned tens_d0, unsigned tens_d1,
                                            unsigned stride_d0, unsigned pad_iv) {
  unsigned long long ga = (unsigned long long)gptr;
  u32x4 g0;
  g0[0] = 1u;                                            // count=1, user mode
  g0[1] = lds_off;                                       // LDS byte offset
  g0[2] = (unsigned)(ga & 0xFFFFFFFFu);
  g0[3] = (unsigned)((ga >> 32) & 0x1FFFFFFu) | (2u << 30);  // addr[56:32], type=2
  i32x8 g1;
  unsigned w0 = (1u << 16)                               // data_size = 1 (2 bytes)
              | (1u << 20)                               // pad_enable
              | (pad_iv << 22)                           // pad_interval
              | (7u << 25);                              // pad_amount = 8 DWORDs
  g1[0] = (int)w0;
  g1[1] = (int)((tens_d0 & 0xFFFFu) << 16);              // tensor_dim0[15:0]
  g1[2] = (int)(((tens_d0 >> 16) & 0xFFFFu) | ((tens_d1 & 0xFFFFu) << 16));
  g1[3] = (int)(((tens_d1 >> 16) & 0xFFFFu) | ((tile_d0 & 0xFFFFu) << 16));
  g1[4] = (int)(tile_d1 & 0xFFFFu);                      // tile_dim1, tile_dim2=0
  g1[5] = (int)stride_d0;                                // tensor_dim0_stride[31:0]
  g1[6] = 0;
  g1[7] = 0;
  i32x4 z4 = {0, 0, 0, 0};                               // groups 2/3 unused (2D)
  i32x8 z8 = {0, 0, 0, 0, 0, 0, 0, 0};
  __builtin_amdgcn_tensor_load_to_lds(g0, g1, z4, z4, z8, 0);
}

// ---------------------------------------------------------------- frag loads
// A-matrix (16x32 bf16, row-major src):
//   lane<16 : row=lane,    K in {0..15}; lane>=16: row=lane-16, K {16..31}
__device__ __forceinline__ v16bf load_a_frag(const bf16* __restrict__ rowptr,
                                             int k0, int half8) {
  v8bf lo = *(const v8bf*)(rowptr + k0 + 2 * half8);
  v8bf hi = *(const v8bf*)(rowptr + k0 + 2 * half8 + 8);
  return __builtin_shufflevector(lo, hi, 0,1,2,3,4,5,6,7,8,9,10,11,12,13,14,15);
}

// ---------------------------------------------------------------- utilities
__global__ void k_zero_u32(unsigned* __restrict__ p, int n) {
  int i = blockIdx.x * 256 + threadIdx.x;
  if (i < n) p[i] = 0u;
}

__global__ void k_convert_w(const float* __restrict__ w, bf16* __restrict__ wt) {
  int i = blockIdx.x * 256 + threadIdx.x;            // 3*C*C
  int t  = i / (Cc * Cc);
  int r  = i % (Cc * Cc);
  int co = r / Cc, ci = r % Cc;
  wt[i] = (bf16)w[(co * Cc + ci) * 3 + t];
}

__global__ void k_build_imageT(const float* __restrict__ img, bf16* __restrict__ it) {
  int i = blockIdx.x * 256 + threadIdx.x;            // Bn*LP*Cc
  int c  = i % Cc;
  int lp = (i / Cc) % LP;
  int b  = i / (Cc * LP);
  float v = (lp == 0 || lp == LP - 1) ? 0.f
            : img[((size_t)b * Cc + c) * Ll + (lp - 1)];
  it[i] = (bf16)v;
}

__global__ void k_conv_small(const float* __restrict__ x, const float* __restrict__ w,
                             const float* __restrict__ bias, bf16* __restrict__ out,
                             int transposed) {
  int i = blockIdx.x * 256 + threadIdx.x;            // Bn*Cc*Ll
  int c, l, b;
  if (transposed) { c = i % Cc; l = (i / Cc) % Ll; b = i / (Cc * Ll); }
  else            { l = i % Ll; c = (i / Ll) % Cc; b = i / (Cc * Ll); }
  const float* xb = x + (size_t)b * Ll;
  float xm = (l > 0)      ? xb[l - 1] : 0.f;
  float x0 = xb[l];
  float xp = (l < Ll - 1) ? xb[l + 1] : 0.f;
  float y = w[c * 3] * xm + w[c * 3 + 1] * x0 + w[c * 3 + 2] * xp + bias[c];
  if (transposed) out[((size_t)b * Ll + l) * Cc + c] = (bf16)y;
  else            out[((size_t)b * Cc + c) * Ll + l] = (bf16)y;
}

// ---------------------------------------------------------------- big conv GEMM
// D[co][l] = sum_{t,ci} W[t][co][ci] * xT[b][l+t][ci] (+bias)
// B^T tile (66 rows x 256) staged once per block via TDM; shared by 8 waves.
// Ping-pong register-pipelined K loop (24 steps, unroll-by-2): fragment sets
// a0/b0 and a1/b1 alternate; each set is written only by loads -> no copies,
// and loads for step k+1/k+2 are in flight behind step k's WMMAs.
// mode 0: bf16 [B][L][C]; 1: bf16 [B][C][L]; 2: f32 [B][L][C]
__global__ void k_conv_big(const bf16* __restrict__ wts, const float* __restrict__ bias,
                           const bf16* __restrict__ xT, void* __restrict__ outp, int mode) {
  __shared__ __align__(64) bf16 smB[66 * BSTR];      // 35,904 B
  int wave = threadIdx.x >> 5, lane = threadIdx.x & 31;
  int b   = blockIdx.z;
  int l0  = blockIdx.x * 64;
  int co0 = (blockIdx.y * 8 + wave) * 16;
  int half8 = (lane >> 4) * 8;
  int k16   = (lane >> 4) * 16;
  int nlan  = lane & 15;

  if (threadIdx.x < 32) {
    tdm_load_2d((unsigned)(unsigned long long)&smB[0],
                xT + ((size_t)b * LP + l0) * Cc,
                /*tile*/ Cc, 66, /*tensor*/ Cc, (unsigned)(LP - l0),
                /*stride*/ Cc, /*pad_iv 128DW*/ 6u);
    __builtin_amdgcn_s_wait_tensorcnt(0);
  }
  __syncthreads();

  const bf16* A0 = wts + (size_t)(co0 + nlan) * Cc;  // + t*Cc*Cc per slice
  auto ldA = [&](int kk) -> v16bf {
    int t = kk >> 3;
    return load_a_frag(A0 + (size_t)t * Cc * Cc, (kk & 7) * 32, half8);
  };
  auto ldB = [&](int kk, int f) -> v16bf {
    int t = kk >> 3;
    return *(const v16bf*)(&smB[(f * 16 + nlan + t) * BSTR + (kk & 7) * 32 + k16]);
  };

  const int NK = 24;                                 // 3 taps x 8 k-steps (even)
  v8f acc[4] = {};
  v16bf a0 = ldA(0), a1;
  v16bf b0[4], b1[4];
  #pragma unroll
  for (int f = 0; f < 4; ++f) b0[f] = ldB(0, f);

  for (int kk = 0; kk < NK; kk += 2) {
    a1 = ldA(kk + 1);
    #pragma unroll
    for (int f = 0; f < 4; ++f) b1[f] = ldB(kk + 1, f);
    #pragma unroll
    for (int f = 0; f < 4; ++f)
      acc[f] = __builtin_amdgcn_wmma_f32_16x16x32_bf16(
          false, a0, false, b0[f], (short)0, acc[f], false, false);
    int k2 = (kk + 2 < NK) ? kk + 2 : kk;            // clamped tail reload
    a0 = ldA(k2);
    #pragma unroll
    for (int f = 0; f < 4; ++f) b0[f] = ldB(k2, f);
    #pragma unroll
    for (int f = 0; f < 4; ++f)
      acc[f] = __builtin_amdgcn_wmma_f32_16x16x32_bf16(
          false, a1, false, b1[f], (short)0, acc[f], false, false);
  }

  float bs[8];
  #pragma unroll
  for (int i = 0; i < 8; ++i) bs[i] = bias[co0 + half8 + i];

  if (mode == 0) {
    bf16* o = (bf16*)outp;
    #pragma unroll
    for (int f = 0; f < 4; ++f) {
      int l = l0 + f * 16 + nlan;
      v8bf v;
      #pragma unroll
      for (int i = 0; i < 8; ++i) v[i] = (bf16)(acc[f][i] + bs[i]);
      *(v8bf*)(o + ((size_t)b * Ll + l) * Cc + co0 + half8) = v;
    }
  } else if (mode == 1) {
    bf16* o = (bf16*)outp;
    #pragma unroll
    for (int f = 0; f < 4; ++f) {
      int l = l0 + f * 16 + nlan;
      #pragma unroll
      for (int i = 0; i < 8; ++i)
        o[((size_t)b * Cc + co0 + half8 + i) * Ll + l] = (bf16)(acc[f][i] + bs[i]);
    }
  } else {
    float* o = (float*)outp;
    #pragma unroll
    for (int f = 0; f < 4; ++f) {
      int l = l0 + f * 16 + nlan;
      v8f v = acc[f];
      #pragma unroll
      for (int i = 0; i < 8; ++i) v[i] += bs[i];
      *(v8f*)(o + ((size_t)b * Ll + l) * Cc + co0 + half8) = v;
    }
  }
}

// ---------------------------------------------------------------- scores GEMM
// D[m][l] = sum_c kT[m][c] * qT[l][c] -> scores[b][l][m] / sqrt(C)
// B^T tile (64 x 256) via TDM; ping-pong pipelined K loop (8 steps).
__global__ void k_scores(const bf16* __restrict__ kT, const bf16* __restrict__ qT,
                         float* __restrict__ sc) {
  __shared__ __align__(64) bf16 smB[64 * BSTR];      // 34,816 B
  int wave = threadIdx.x >> 5, lane = threadIdx.x & 31;
  int b  = blockIdx.z;
  int l0 = blockIdx.x * 64;
  int m0 = (blockIdx.y * 8 + wave) * 16;
  int half8 = (lane >> 4) * 8, k16 = (lane >> 4) * 16, nlan = lane & 15;
  const bf16* A = kT + (size_t)b * Ll * Cc + (size_t)(m0 + nlan) * Cc;

  if (threadIdx.x < 32) {
    tdm_load_2d((unsigned)(unsigned long long)&smB[0],
                qT + ((size_t)b * Ll + l0) * Cc,
                Cc, 64, Cc, (unsigned)(Ll - l0), Cc, 6u);
    __builtin_amdgcn_s_wait_tensorcnt(0);
  }
  __syncthreads();

  auto ldB = [&](int kk, int f) -> v16bf {
    return *(const v16bf*)(&smB[(f * 16 + nlan) * BSTR + kk * 32 + k16]);
  };

  const int NK = 8;
  v8f acc[4] = {};
  v16bf a0 = load_a_frag(A, 0, half8), a1;
  v16bf b0[4], b1[4];
  #pragma unroll
  for (int f = 0; f < 4; ++f) b0[f] = ldB(0, f);

  for (int kk = 0; kk < NK; kk += 2) {
    a1 = load_a_frag(A, (kk + 1) * 32, half8);
    #pragma unroll
    for (int f = 0; f < 4; ++f) b1[f] = ldB(kk + 1, f);
    #pragma unroll
    for (int f = 0; f < 4; ++f)
      acc[f] = __builtin_amdgcn_wmma_f32_16x16x32_bf16(
          false, a0, false, b0[f], (short)0, acc[f], false, false);
    int k2 = (kk + 2 < NK) ? kk + 2 : kk;
    a0 = load_a_frag(A, k2 * 32, half8);
    #pragma unroll
    for (int f = 0; f < 4; ++f) b0[f] = ldB(k2, f);
    #pragma unroll
    for (int f = 0; f < 4; ++f)
      acc[f] = __builtin_amdgcn_wmma_f32_16x16x32_bf16(
          false, a1, false, b1[f], (short)0, acc[f], false, false);
  }

  const float scale = 0.0625f;                        // 1/sqrt(256)
  #pragma unroll
  for (int f = 0; f < 4; ++f) {
    int l = l0 + f * 16 + nlan;
    v8f v = acc[f];
    #pragma unroll
    for (int i = 0; i < 8; ++i) v[i] *= scale;
    *(v8f*)(sc + ((size_t)b * Ll + l) * Ll + m0 + half8) = v;
  }
}

// ---------------------------------------------------------------- softmax rows
__global__ void k_softmax(const float* __restrict__ s, bf16* __restrict__ p) {
  __shared__ float red[256];
  int row = blockIdx.x, t = threadIdx.x;
  const float* sr = s + (size_t)row * Ll;
  float v0 = sr[t], v1 = sr[t + 256], v2 = sr[t + 512], v3 = sr[t + 768];
  float m = fmaxf(fmaxf(v0, v1), fmaxf(v2, v3));
  red[t] = m; __syncthreads();
  for (int o = 128; o > 0; o >>= 1) { if (t < o) red[t] = fmaxf(red[t], red[t + o]); __syncthreads(); }
  m = red[0]; __syncthreads();
  float e0 = __expf(v0 - m), e1 = __expf(v1 - m), e2 = __expf(v2 - m), e3 = __expf(v3 - m);
  red[t] = e0 + e1 + e2 + e3; __syncthreads();
  for (int o = 128; o > 0; o >>= 1) { if (t < o) red[t] += red[t + o]; __syncthreads(); }
  float inv = 1.0f / red[0];
  bf16* pr = p + (size_t)row * Ll;
  pr[t]       = (bf16)(e0 * inv);
  pr[t + 256] = (bf16)(e1 * inv);
  pr[t + 512] = (bf16)(e2 * inv);
  pr[t + 768] = (bf16)(e3 * inv);
}

// ---------------------------------------------------------------- ctx GEMM
// D[c][l] = sum_m v[c][m] * probs[l][m] -> ctxT[b][l+1][c]
// K=1024: 8 chunks of 128, double-buffered TDM overlaps the next chunk's DMA
// with this chunk's compute; inside a chunk the 4 K-steps are ping-pong pipelined.
__global__ void k_ctx(const bf16* __restrict__ vv, const bf16* __restrict__ pr,
                      bf16* __restrict__ ctxT) {
  __shared__ __align__(64) bf16 smB[2][64 * CTX_STR];  // 2 x 18,432 B
  int wave = threadIdx.x >> 5, lane = threadIdx.x & 31;
  int b  = blockIdx.z;
  int l0 = blockIdx.x * 64;
  int c0 = (blockIdx.y * 8 + wave) * 16;
  int half8 = (lane >> 4) * 8, k16 = (lane >> 4) * 16, nlan = lane & 15;
  const bf16* A  = vv + (size_t)b * Cc * Ll + (size_t)(c0 + nlan) * Ll;
  const bf16* Bt = pr + (size_t)b * Ll * Ll + (size_t)l0 * Ll;
  const int NCH = Ll / CTX_CH;                         // 8

  if (threadIdx.x < 32)
    tdm_load_2d((unsigned)(unsigned long long)&smB[0][0], Bt,
                CTX_CH, 64, Ll, (unsigned)(Ll - l0), Ll, /*pad_iv 64DW*/ 5u);

  v8f acc[4] = {};
  for (int ch = 0; ch < NCH; ++ch) {
    if (threadIdx.x < 32) __builtin_amdgcn_s_wait_tensorcnt(0);
    __syncthreads();                                   // chunk ch ready; buf (ch+1)&1 free
    if (ch + 1 < NCH && threadIdx.x < 32)
      tdm_load_2d((unsigned)(unsigned long long)&smB[(ch + 1) & 1][0],
                  Bt + (ch + 1) * CTX_CH,
                  CTX_CH, 64, (unsigned)(Ll - (ch + 1) * CTX_CH),
                  (unsigned)(Ll - l0), Ll, 5u);
    const bf16* sm = &smB[ch & 1][0];
    auto ldB = [&](int kk, int f) -> v16bf {
      return *(const v16bf*)(sm + (f * 16 + nlan) * CTX_STR + kk * 32 + k16);
    };
    const int NK = CTX_CH / 32;                        // 4 (even)
    v16bf a0 = load_a_frag(A, ch * CTX_CH, half8), a1;
    v16bf b0[4], b1[4];
    #pragma unroll
    for (int f = 0; f < 4; ++f) b0[f] = ldB(0, f);
    #pragma unroll
    for (int kk = 0; kk < NK; kk += 2) {
      a1 = load_a_frag(A, ch * CTX_CH + (kk + 1) * 32, half8);
      #pragma unroll
      for (int f = 0; f < 4; ++f) b1[f] = ldB(kk + 1, f);
      #pragma unroll
      for (int f = 0; f < 4; ++f)
        acc[f] = __builtin_amdgcn_wmma_f32_16x16x32_bf16(
            false, a0, false, b0[f], (short)0, acc[f], false, false);
      int k2 = (kk + 2 < NK) ? kk + 2 : kk;
      a0 = load_a_frag(A, ch * CTX_CH + k2 * 32, half8);
      #pragma unroll
      for (int f = 0; f < 4; ++f) b0[f] = ldB(k2, f);
      #pragma unroll
      for (int f = 0; f < 4; ++f)
        acc[f] = __builtin_amdgcn_wmma_f32_16x16x32_bf16(
            false, a1, false, b1[f], (short)0, acc[f], false, false);
    }
    __syncthreads();                                   // all done reading buf ch&1
  }
  #pragma unroll
  for (int f = 0; f < 4; ++f) {
    int l = l0 + f * 16 + nlan;
    v8bf v;
    #pragma unroll
    for (int i = 0; i < 8; ++i) v[i] = (bf16)acc[f][i];
    *(v8bf*)(ctxT + ((size_t)b * LP + l + 1) * Cc + c0 + half8) = v;
  }
}

// ---------------------------------------------------------------- BN
__global__ void k_bn_stats(const float* __restrict__ yT, float* __restrict__ stats) {
  int c = threadIdx.x;
  int row0 = blockIdx.x * 256;
  float s = 0.f, s2 = 0.f;
  for (int r = 0; r < 256; ++r) {
    float v = yT[(size_t)(row0 + r) * Cc + c];
    s += v; s2 += v * v;
  }
  atomicAdd(&stats[c], s);
  atomicAdd(&stats[Cc + c], s2);
}

__global__ void k_bn_apply(const float* __restrict__ yT, const float* __restrict__ stats,
                           const float* __restrict__ g, const float* __restrict__ be,
                           const float* __restrict__ img, float* __restrict__ out,
                           int chanOff) {
  int i = blockIdx.x * 256 + threadIdx.x;            // Bn*Cc*Ll
  int l = i % Ll, c = (i / Ll) % Cc, b = i / (Cc * Ll);
  float y    = yT[((size_t)b * Ll + l) * Cc + c];
  float mean = stats[c] * (1.f / (float)NROW);
  float var  = stats[Cc + c] * (1.f / (float)NROW) - mean * mean;
  float o = (y - mean) * rsqrtf(var + EPSV) * g[c] + be[c]
            + img[((size_t)b * Cc + c) * Ll + l];
  out[((size_t)b * 2 * Cc + chanOff + c) * Ll + l] = o;
}

// ---------------------------------------------------------------- host
extern "C" void kernel_launch(void* const* d_in, const int* in_sizes, int n_in,
                              void* d_out, int out_size, void* d_ws, size_t ws_size,
                              hipStream_t stream) {
  (void)in_sizes; (void)n_in; (void)out_size; (void)ws_size;
  const float* image    = (const float*)d_in[0];
  const float* clinical = (const float*)d_in[1];
  const float* a_qw = (const float*)d_in[2];  const float* a_qb = (const float*)d_in[3];
  const float* a_kw = (const float*)d_in[4];  const float* a_kb = (const float*)d_in[5];
  const float* a_vw = (const float*)d_in[6];  const float* a_vb = (const float*)d_in[7];
  const float* a_ow = (const float*)d_in[8];  const float* a_ob = (const float*)d_in[9];
  const float* a_g  = (const float*)d_in[10]; const float* a_be = (const float*)d_in[11];
  const float* b_qw = (const float*)d_in[12]; const float* b_qb = (const float*)d_in[13];
  const float* b_kw = (const float*)d_in[14]; const float* b_kb = (const float*)d_in[15];
  const float* b_vw = (const float*)d_in[16]; const float* b_vb = (const float*)d_in[17];
  const float* b_ow = (const float*)d_in[18]; const float* b_ob = (const float*)d_in[19];
  const float* b_g  = (const float*)d_in[20]; const float* b_be = (const float*)d_in[21];
  float* out = (float*)d_out;

  char* ws = (char*)d_ws;
  bf16*  imageT = (bf16*)ws;                 ws += (size_t)Bn * LP * Cc * 2;
  bf16*  wsl0   = (bf16*)ws;                 ws += (size_t)3 * Cc * Cc * 2;
  bf16*  wsl1   = (bf16*)ws;                 ws += (size_t)3 * Cc * Cc * 2;
  bf16*  wsl2   = (bf16*)ws;                 ws += (size_t)3 * Cc * Cc * 2;
  bf16*  bufKT  = (bf16*)ws;                 ws += (size_t)Bn * Ll * Cc * 2;
  bf16*  bufQT  = (bf16*)ws;                 ws += (size_t)Bn * Ll * Cc * 2;
  bf16*  bufV   = (bf16*)ws;                 ws += (size_t)Bn * Cc * Ll * 2;
  float* scores = (float*)ws;                ws += (size_t)Bn * Ll * Ll * 4;
  bf16*  probs  = (bf16*)ws;                 ws += (size_t)Bn * Ll * Ll * 2;
  bf16*  ctxT   = (bf16*)ws;                 ws += (size_t)Bn * LP * Cc * 2;
  float* yT     = (float*)ws;                ws += (size_t)Bn * Ll * Cc * 4;
  float* stats  = (float*)ws;                ws += 2 * Cc * 4;

  const int TPB = 256;
  dim3 gGemm(Ll / 64, Cc / 128, Bn);
  dim3 gSc(Ll / 64, Ll / 16 / 8, Bn);
  dim3 gCtx(Ll / 64, Cc / 16 / 8, Bn);
  int nElem  = Bn * Cc * Ll;
  int nImT   = Bn * LP * Cc;
  int nW     = 3 * Cc * Cc;
  int ctxW32 = (int)((size_t)Bn * LP * Cc * 2 / 4);

  k_build_imageT<<<(nImT + TPB - 1) / TPB, TPB, 0, stream>>>(image, imageT);

  for (int br = 0; br < 2; ++br) {
    const float *qw, *qb, *kw, *kb, *vw, *vb, *ow, *ob, *g, *be;
    if (br == 0) { qw=a_qw; qb=a_qb; kw=a_kw; kb=a_kb; vw=a_vw; vb=a_vb; ow=a_ow; ob=a_ob; g=a_g; be=a_be; }
    else         { qw=b_qw; qb=b_qb; kw=b_kw; kb=b_kb; vw=b_vw; vb=b_vb; ow=b_ow; ob=b_ob; g=b_g; be=b_be; }

    if (br == 0) {
      k_convert_w<<<nW / TPB, TPB, 0, stream>>>(qw, wsl0);
      k_convert_w<<<nW / TPB, TPB, 0, stream>>>(ow, wsl1);
      k_conv_big<<<gGemm, TPB, 0, stream>>>(wsl0, qb, imageT, (void*)bufQT, 0);
      k_conv_small<<<nElem / TPB, TPB, 0, stream>>>(clinical, kw, kb, bufKT, 1);
      k_conv_small<<<nElem / TPB, TPB, 0, stream>>>(clinical, vw, vb, bufV, 0);
    } else {
      k_convert_w<<<nW / TPB, TPB, 0, stream>>>(kw, wsl0);
      k_convert_w<<<nW / TPB, TPB, 0, stream>>>(vw, wsl1);
      k_convert_w<<<nW / TPB, TPB, 0, stream>>>(ow, wsl2);
      k_conv_small<<<nElem / TPB, TPB, 0, stream>>>(clinical, qw, qb, bufQT, 1);
      k_conv_big<<<gGemm, TPB, 0, stream>>>(wsl0, kb, imageT, (void*)bufKT, 0);
      k_conv_big<<<gGemm, TPB, 0, stream>>>(wsl1, vb, imageT, (void*)bufV, 1);
    }

    k_scores<<<gSc, TPB, 0, stream>>>(bufKT, bufQT, scores);
    k_softmax<<<NROW, TPB, 0, stream>>>(scores, probs);

    k_zero_u32<<<(ctxW32 + TPB - 1) / TPB, TPB, 0, stream>>>((unsigned*)ctxT, ctxW32);
    k_ctx<<<gCtx, TPB, 0, stream>>>(bufV, probs, ctxT);

    const bf16* owbuf = (br == 0) ? wsl1 : wsl2;
    k_conv_big<<<gGemm, TPB, 0, stream>>>(owbuf, ob, ctxT, (void*)yT, 2);

    k_zero_u32<<<2, TPB, 0, stream>>>((unsigned*)stats, 2 * Cc);
    k_bn_stats<<<NROW / 256, TPB, 0, stream>>>(yT, stats);
    k_bn_apply<<<nElem / TPB, TPB, 0, stream>>>(yT, stats, g, be, image, out,
                                                br == 0 ? 0 : Cc);
  }
}
